// MultiAutoEncoder_79783312490761
// MI455X (gfx1250) — compile-verified, matrix operand order checked by reading, not run on hardware
//
#include <hip/hip_runtime.h>
#include <hip/hip_bf16.h>

typedef __attribute__((ext_vector_type(16))) __bf16 v16bf;
typedef __attribute__((ext_vector_type(8)))  __bf16 v8bf;
typedef __attribute__((ext_vector_type(4)))  __bf16 v4bf;
typedef __attribute__((ext_vector_type(8)))  float  v8f;

#define NCELLS 10000
#define HDIM   256
#define KHOPS  10

// ---------------- WMMA GEMM: C = act(A[M,Kd] @ B[Kd,Nc] + bias) -------------
// 512 threads (16 waves), block tile 128(M) x 256(N), K-tile 64, double-buffered
#define BM  128
#define BN  256
#define BKT 64
#define LDT 72   // padded LDS stride (elements); 144B rows keep 16B alignment

__device__ __forceinline__ v16bf load_frag(const __bf16* base, int lane) {
  // striped dim (M for A, N for B^T) = lane&15
  // lanes 0-15 -> K {0..7,16..23}, lanes 16-31 -> K {8..15,24..31}
  int idx = lane & 15;
  int hi  = (lane >> 4) << 3;
  const __bf16* p = base + idx * LDT;
  v8bf lo  = *(const v8bf*)(p + hi);
  v8bf hi2 = *(const v8bf*)(p + 16 + hi);
  return __builtin_shufflevector(lo, hi2, 0,1,2,3,4,5,6,7,8,9,10,11,12,13,14,15);
}

__device__ __forceinline__ v4bf cvt4(float4 f) {
  v4bf r = { (__bf16)f.x, (__bf16)f.y, (__bf16)f.z, (__bf16)f.w };
  return r;
}

struct TileRegs { v4bf a[4]; v4bf b[8]; };

__device__ __forceinline__ void gload_tile(TileRegs& r,
    const float* __restrict__ A, int lda,
    const float* __restrict__ B, int ldb,
    int bm0, int bn0, int k0, int M, int Nc, int Kd, int t)
{
  // A tile: 128x64 = 2048 float4 slots / 512 threads = 4
#pragma unroll
  for (int it = 0; it < 4; ++it) {
    int s = t + it * 512;
    int row = s >> 4, c4 = (s & 15) << 2;
    int gm = bm0 + row, gk = k0 + c4;
    float4 f = {0.f, 0.f, 0.f, 0.f};
    if (gm < M) {
      const float* p = A + (size_t)gm * lda;
      if (gk + 3 < Kd) f = *(const float4*)(p + gk);
      else {
        if (gk + 0 < Kd) f.x = p[gk + 0];
        if (gk + 1 < Kd) f.y = p[gk + 1];
        if (gk + 2 < Kd) f.z = p[gk + 2];
        if (gk + 3 < Kd) f.w = p[gk + 3];
      }
    }
    r.a[it] = cvt4(f);
  }
  // B tile: 64x256 = 4096 float4 slots / 512 threads = 8
#pragma unroll
  for (int it = 0; it < 8; ++it) {
    int s = t + it * 512;
    int kr = s >> 6, n4 = (s & 63) << 2;
    int gk = k0 + kr, gn = bn0 + n4;
    float4 f = {0.f, 0.f, 0.f, 0.f};
    if (gk < Kd) {
      const float* p = B + (size_t)gk * ldb;
      if (gn + 3 < Nc) f = *(const float4*)(p + gn);
      else {
        if (gn + 0 < Nc) f.x = p[gn + 0];
        if (gn + 1 < Nc) f.y = p[gn + 1];
        if (gn + 2 < Nc) f.z = p[gn + 2];
        if (gn + 3 < Nc) f.w = p[gn + 3];
      }
    }
    r.b[it] = cvt4(f);
  }
}

__device__ __forceinline__ void store_tile(const TileRegs& r,
                                           __bf16* dA, __bf16* dB, int t) {
#pragma unroll
  for (int it = 0; it < 4; ++it) {
    int s = t + it * 512;
    int row = s >> 4, c4 = (s & 15) << 2;
    *(v4bf*)&dA[row * LDT + c4] = r.a[it];
  }
#pragma unroll
  for (int it = 0; it < 8; ++it) {
    int s = t + it * 512;
    int kr = s >> 6, n4 = (s & 63) << 2;
    dB[(n4 + 0) * LDT + kr] = r.b[it][0];
    dB[(n4 + 1) * LDT + kr] = r.b[it][1];
    dB[(n4 + 2) * LDT + kr] = r.b[it][2];
    dB[(n4 + 3) * LDT + kr] = r.b[it][3];
  }
}

__global__ __launch_bounds__(512)
void gemm_wmma(const float* __restrict__ A, int lda,
               const float* __restrict__ B, int ldb,
               const float* __restrict__ bias,
               float* __restrict__ C, int ldc,
               int M, int Nc, int Kd, int relu)
{
  __shared__ __bf16 sA[2][BM * LDT];   // [m][k]
  __shared__ __bf16 sB[2][BN * LDT];   // transposed: [n][k]

  const int t    = threadIdx.x;
  const int lane = t & 31;
  const int wid  = t >> 5;            // 0..15
  const int wm   = (wid >> 2) * 32;   // 0,32,64,96
  const int wn   = (wid & 3) * 64;    // 0,64,128,192
  const int bm0  = blockIdx.y * BM;
  const int bn0  = blockIdx.x * BN;

  v8f acc[2][4];
#pragma unroll
  for (int i = 0; i < 2; ++i)
#pragma unroll
    for (int j = 0; j < 4; ++j)
#pragma unroll
      for (int e = 0; e < 8; ++e) acc[i][j][e] = 0.0f;

  const int T = (Kd + BKT - 1) / BKT;
  TileRegs rg;
  gload_tile(rg, A, lda, B, ldb, bm0, bn0, 0, M, Nc, Kd, t);
  store_tile(rg, sA[0], sB[0], t);
  __syncthreads();

  for (int tt = 0; tt < T; ++tt) {
    const bool more = (tt + 1) < T;
    if (more)  // prefetch next tile into registers while this tile computes
      gload_tile(rg, A, lda, B, ldb, bm0, bn0, (tt + 1) * BKT, M, Nc, Kd, t);

    const __bf16* cA = sA[tt & 1];
    const __bf16* cB = sB[tt & 1];
#pragma unroll
    for (int kk = 0; kk < BKT; kk += 32) {
      v16bf af[2], bfg[4];
#pragma unroll
      for (int mi = 0; mi < 2; ++mi)
        af[mi] = load_frag(&cA[(wm + mi * 16) * LDT + kk], lane);
#pragma unroll
      for (int ni = 0; ni < 4; ++ni)
        bfg[ni] = load_frag(&cB[(wn + ni * 16) * LDT + kk], lane);
#pragma unroll
      for (int mi = 0; mi < 2; ++mi)
#pragma unroll
        for (int ni = 0; ni < 4; ++ni)
          acc[mi][ni] = __builtin_amdgcn_wmma_f32_16x16x32_bf16(
              false, af[mi], false, bfg[ni], (short)0, acc[mi][ni], false, false);
    }
    if (more) {
      store_tile(rg, sA[(tt + 1) & 1], sB[(tt + 1) & 1], t);
      __syncthreads();
    }
  }

  // epilogue: lanes 0-15 -> M=r, N=lane; lanes 16-31 -> M=8+r, N=lane-16
  const int r_hi = (lane >> 4) << 3;
  const int n_in = lane & 15;
#pragma unroll
  for (int mi = 0; mi < 2; ++mi)
#pragma unroll
    for (int ni = 0; ni < 4; ++ni) {
      int gn = bn0 + wn + ni * 16 + n_in;
      if (gn >= Nc) continue;
      float bv = bias ? bias[gn] : 0.f;
#pragma unroll
      for (int r = 0; r < 8; ++r) {
        int gm = bm0 + wm + mi * 16 + r_hi + r;
        if (gm < M) {
          float v = acc[mi][ni][r] + bv;
          if (relu) v = fmaxf(v, 0.f);
          C[(size_t)gm * ldc + gn] = v;
        }
      }
    }
}

// ---------------- small helpers -------------------------------------------
// Wi[2][D][256] -> dst[D][512] so both orders share one GEMM pass over X
__global__ void pack_wi(const float* __restrict__ Wi, float* __restrict__ dst, int D) {
  int i = blockIdx.x * blockDim.x + threadIdx.x;
  if (i < D * 512) {
    int d = i >> 9, c = i & 511;
    int o = c >> 8, h = c & 255;
    dst[i] = Wi[((size_t)o * D + d) * 256 + h];
  }
}

__global__ void softmax_fw_k(const float* __restrict__ fwr,
                             const float* __restrict__ fwa,
                             float* __restrict__ out) {
  int r = threadIdx.x;
  if (r >= 4) return;
  const float* src = (r < 2) ? (fwr + r * (KHOPS + 1)) : (fwa + (r - 2) * (KHOPS + 1));
  float mx = -1e30f;
  for (int j = 0; j <= KHOPS; ++j) mx = fmaxf(mx, src[j]);
  float e[KHOPS + 1], s = 0.f;
  for (int j = 0; j <= KHOPS; ++j) { e[j] = __expf(src[j] - mx); s += e[j]; }
  float inv = 1.f / s;
  for (int j = 0; j <= KHOPS; ++j) out[r * (KHOPS + 1) + j] = e[j] * inv;
}

__global__ void zero_f32(float* __restrict__ p, int n) {
  int i = blockIdx.x * blockDim.x + threadIdx.x;
  if (i < n) p[i] = 0.f;
}

__global__ void scale_strided(float* __restrict__ dst, int dstStride,
                              const float* __restrict__ src, int srcStride,
                              const float* __restrict__ wp, int widx, int n) {
  int i = blockIdx.x * blockDim.x + threadIdx.x;
  if (i < n) {
    float w = wp[widx];
    dst[(size_t)(i >> 8) * dstStride + (i & 255)] =
        w * src[(size_t)(i >> 8) * srcStride + (i & 255)];
  }
}

__global__ void axpy_strided(float* __restrict__ dst, int dstStride,
                             const float* __restrict__ src,
                             const float* __restrict__ wp, int widx, int n) {
  int i = blockIdx.x * blockDim.x + threadIdx.x;
  if (i < n) {
    float w = wp[widx];
    dst[(size_t)(i >> 8) * dstStride + (i & 255)] += w * src[i];
  }
}

// 64 threads per edge, 4 floats each: y[row] += val * x[col]; x may be strided
__global__ __launch_bounds__(256)
void spmm_scatter(const int* __restrict__ row, const int* __restrict__ col,
                  const float* __restrict__ val,
                  const float* __restrict__ x, int ldx,
                  float* __restrict__ y, int E) {
  unsigned tid = blockIdx.x * 256u + threadIdx.x;
  unsigned e = tid >> 6;
  unsigned c = (tid & 63u) << 2;
  if (e >= (unsigned)E) return;
  int r = row[e], cl = col[e];
  float v = val[e];
  float4 xv = *(const float4*)(x + (size_t)cl * ldx + c);
  float* yp = y + (size_t)r * HDIM + c;
  atomicAdd(yp + 0, v * xv.x);
  atomicAdd(yp + 1, v * xv.y);
  atomicAdd(yp + 2, v * xv.z);
  atomicAdd(yp + 3, v * xv.w);
}

// one wave32 per row: logits, softmax over 2, fuse
__global__ __launch_bounds__(256)
void attention_fuse(const float* __restrict__ z_rna, const float* __restrict__ z_atac,
                    const float* __restrict__ Wa, const float* __restrict__ ba,
                    float* __restrict__ z, float* __restrict__ weight, int Nrows) {
  int gw = (blockIdx.x * blockDim.x + threadIdx.x) >> 5;
  int lane = threadIdx.x & 31;
  if (gw >= Nrows) return;
  const float* zr = z_rna + (size_t)gw * HDIM;
  const float* za = z_atac + (size_t)gw * HDIM;
  float a0 = 0.f, a1 = 0.f;
  for (int j = lane; j < HDIM; j += 32) {
    float v = zr[j];
    a0 += v * Wa[j * 2 + 0];
    a1 += v * Wa[j * 2 + 1];
    float v2 = za[j];
    a0 += v2 * Wa[(HDIM + j) * 2 + 0];
    a1 += v2 * Wa[(HDIM + j) * 2 + 1];
  }
#pragma unroll
  for (int off = 16; off > 0; off >>= 1) {
    a0 += __shfl_down(a0, off, 32);
    a1 += __shfl_down(a1, off, 32);
  }
  a0 = __shfl(a0, 0, 32) + ba[0];
  a1 = __shfl(a1, 0, 32) + ba[1];
  float mx = fmaxf(a0, a1);
  float e0 = __expf(a0 - mx), e1 = __expf(a1 - mx);
  float inv = 1.f / (e0 + e1);
  float w0 = e0 * inv, w1 = e1 * inv;
  if (lane == 0) {
    weight[(size_t)gw * 2 + 0] = w0;
    weight[(size_t)gw * 2 + 1] = w1;
  }
  for (int j = lane; j < HDIM; j += 32)
    z[(size_t)gw * HDIM + j] = w0 * zr[j] + w1 * za[j];
}

// ---------------- host orchestration --------------------------------------
extern "C" void kernel_launch(void* const* d_in, const int* in_sizes, int n_in,
                              void* d_out, int out_size, void* d_ws, size_t ws_size,
                              hipStream_t stream) {
  (void)n_in; (void)out_size; (void)ws_size;
  const float* X_rna   = (const float*)d_in[0];
  const float* X_atac  = (const float*)d_in[1];
  const int*   row1    = (const int*)d_in[2];
  const int*   col1    = (const int*)d_in[3];
  const float* val1    = (const float*)d_in[4];
  const int*   row2    = (const int*)d_in[5];
  const int*   col2    = (const int*)d_in[6];
  const float* val2    = (const float*)d_in[7];
  const float* Wi_rna  = (const float*)d_in[8];
  const float* bi_rna  = (const float*)d_in[9];
  const float* fW_rna  = (const float*)d_in[10];
  const float* Wo_rna  = (const float*)d_in[11];
  const float* bo_rna  = (const float*)d_in[12];
  const float* Wi_atac = (const float*)d_in[13];
  const float* bi_atac = (const float*)d_in[14];
  const float* fW_atac = (const float*)d_in[15];
  const float* Wo_atac = (const float*)d_in[16];
  const float* bo_atac = (const float*)d_in[17];
  const float* Wa      = (const float*)d_in[18];
  const float* ba      = (const float*)d_in[19];
  const float* Wd1_rna = (const float*)d_in[20];
  const float* bd1_rna = (const float*)d_in[21];
  const float* Wd2_rna = (const float*)d_in[22];
  const float* bd2_rna = (const float*)d_in[23];
  const float* Wd1_atac = (const float*)d_in[24];
  const float* bd1_atac = (const float*)d_in[25];
  const float* Wd2_atac = (const float*)d_in[26];
  const float* bd2_atac = (const float*)d_in[27];
  const int E = in_sizes[2];

  float* out      = (float*)d_out;
  float* z        = out;
  float* z_rna    = out + (size_t)NCELLS * 256;
  float* z_atac   = out + (size_t)NCELLS * 512;
  float* weight   = out + (size_t)NCELLS * 768;
  float* rna_rec  = weight + (size_t)NCELLS * 2;
  float* atac_rec = rna_rec + (size_t)NCELLS * 2000;

  float* ws        = (float*)d_ws;
  float* fw_soft   = ws;                                   // 44 -> 64
  float* zcat_rna  = ws + 64;                              // N*512
  float* zcat_atac = zcat_rna  + (size_t)NCELLS * 512;     // N*512
  float* hbuf2     = zcat_atac + (size_t)NCELLS * 512;     // N*512 (both orders)
  float* pbufA     = hbuf2     + (size_t)NCELLS * 512;     // N*256
  float* pbufB     = pbufA     + (size_t)NCELLS * 256;     // N*256
  float* wipack    = pbufB     + (size_t)NCELLS * 256;     // 5000*512 max

  dim3 blk256(256);
  dim3 blk512(512);
  const int NE = NCELLS * HDIM;
  const int eb = (NE + 255) / 256;
  const unsigned spmmBlocks = ((unsigned)E * 64u + 255u) / 256u;

  softmax_fw_k<<<1, 64, 0, stream>>>(fW_rna, fW_atac, fw_soft);

  const int*   rows[2] = {row1, row2};
  const int*   cols[2] = {col1, col2};
  const float* vals[2] = {val1, val2};

  struct Mod {
    const float* X; int D;
    const float* Wi; const float* bi; int fwBase;
    const float* Wo; const float* bo;
    float* zcat; float* zout;
  };
  Mod mods[2] = {
    {X_rna,  2000, Wi_rna,  bi_rna,  0, Wo_rna,  bo_rna,  zcat_rna,  z_rna},
    {X_atac, 5000, Wi_atac, bi_atac, 2, Wo_atac, bo_atac, zcat_atac, z_atac},
  };

  const dim3 gMy(1, (NCELLS + BM - 1) / BM);   // N=256 GEMMs
  const dim3 gH2(2, (NCELLS + BM - 1) / BM);   // N=512 GEMMs

  for (int m = 0; m < 2; ++m) {
    const int D = mods[m].D;
    // pack Wi into [D,512] (both orders side-by-side); bi is already flat [512]
    pack_wi<<<(D * 512 + 255) / 256, blk256, 0, stream>>>(mods[m].Wi, wipack, D);
    // h(both orders) = X @ WiPack + bi  -> hbuf2 [N,512], X streamed once
    gemm_wmma<<<gH2, blk512, 0, stream>>>(mods[m].X, D, wipack, 512,
                                          mods[m].bi, hbuf2, 512,
                                          NCELLS, 512, D, 0);

    for (int o = 0; o < 2; ++o) {
      int widx = (mods[m].fwBase + o) * (KHOPS + 1);
      // acc = fw[0]*h  (strided read from hbuf2, strided write into concat)
      scale_strided<<<eb, blk256, 0, stream>>>(mods[m].zcat + o * 256, 512,
                                               hbuf2 + o * 256, 512,
                                               fw_soft, widx + 0, NE);
      const float* ping = hbuf2 + o * 256;
      int ldx = 512;
      float* pong = pbufA;
      for (int k = 1; k <= KHOPS; ++k) {
        zero_f32<<<eb, blk256, 0, stream>>>(pong, NE);
        spmm_scatter<<<spmmBlocks, blk256, 0, stream>>>(
            rows[o], cols[o], vals[o], ping, ldx, pong, E);
        axpy_strided<<<eb, blk256, 0, stream>>>(mods[m].zcat + o * 256, 512,
                                                pong, fw_soft, widx + k, NE);
        ping = pong; ldx = HDIM;
        pong = (ping == pbufA) ? pbufB : pbufA;
      }
    }
    // z_m = concat(z1,z2) @ Wo + bo
    gemm_wmma<<<gMy, blk512, 0, stream>>>(mods[m].zcat, 512, mods[m].Wo, 256,
                                          mods[m].bo, mods[m].zout, 256,
                                          NCELLS, 256, 512, 0);
  }

  // attention fusion -> z, weight
  attention_fuse<<<(NCELLS * 32 + 255) / 256, blk256, 0, stream>>>(
      z_rna, z_atac, Wa, ba, z, weight, NCELLS);

  // decoders (d1 buffer reuses pbufA)
  gemm_wmma<<<gMy, blk512, 0, stream>>>(z, 256, Wd1_rna, 256, bd1_rna,
                                        pbufA, 256, NCELLS, 256, 256, 1);
  gemm_wmma<<<dim3((2000 + BN - 1) / BN, (NCELLS + BM - 1) / BM), blk512, 0, stream>>>(
      pbufA, 256, Wd2_rna, 2000, bd2_rna, rna_rec, 2000, NCELLS, 2000, 256, 0);

  gemm_wmma<<<gMy, blk512, 0, stream>>>(z, 256, Wd1_atac, 256, bd1_atac,
                                        pbufA, 256, NCELLS, 256, 256, 1);
  gemm_wmma<<<dim3((5000 + BN - 1) / BN, (NCELLS + BM - 1) / BM), blk512, 0, stream>>>(
      pbufA, 256, Wd2_atac, 5000, bd2_atac, atac_rec, 5000, NCELLS, 5000, 256, 0);
}
